// EncoderLSTM_22411139351142
// MI455X (gfx1250) — matrix operand order repaired and checked by reading, not compiled, not run on hardware
//
#include <hip/hip_runtime.h>

typedef __bf16 bf16;
typedef __bf16 v16bf __attribute__((ext_vector_type(16)));
typedef float  v8f   __attribute__((ext_vector_type(8)));

#define T_LEN 512
#define B_SZ  256
#define D_IN  32
#define H_DIM 512
#define G_DIM 2048        // 4*H
#define KX    64          // padded input K (32 y + 1 t + 1 bias + pad)
#define KCAT  576         // KX + H_DIM : unified A strip
#define KT_ALL 18         // KCAT / 32 K-tiles
#define BT    16          // batch rows per workgroup (one WMMA M tile)
#define NWAVE 16          // waves per workgroup
#define NT_W  8           // 16x16 N-tiles per wave (8*16 = 128 packed cols)

#define JSTRIDE (KT_ALL * 32 * 16)   // 9216 elements between consecutive N-tiles

union BFVec { v16bf v; uint4 q[2]; };

__device__ __forceinline__ float sigm_(float x) { return 1.0f / (1.0f + __expf(-x)); }
__device__ __forceinline__ float tanh_(float x) { return 1.0f - 2.0f / (1.0f + __expf(2.0f * x)); }

// ---------------------------------------------------------------------------
// Kernel 0: pack [W_ihp | W_hh] -> one bf16 buffer in WMMA B-operand layout.
// Packed gate column p (0..2047) maps to gate row g = (p&3)*512 + (p>>2)
// (i,f,g,o interleaved per h-column so updates are wave-local).
// Layout: idx = ((nt*18 + kt)*32 + L)*16 + j
//   N = nt*16 + (L&15),  K = kt*32 + (L>>4)*16 + j   (K in 0..575)
//   K<33 -> W_ih, K==33 -> b_ih+b_hh, 33<K<64 -> 0, K>=64 -> W_hh[g][K-64]
// ---------------------------------------------------------------------------
__global__ void pack_weights(const float* __restrict__ W_ih,
                             const float* __restrict__ W_hh,
                             const float* __restrict__ b_ih,
                             const float* __restrict__ b_hh,
                             bf16* __restrict__ w_pk)
{
    int id = blockIdx.x * blockDim.x + threadIdx.x;
    if (id >= G_DIM * KCAT) return;
    int j    = id & 15;
    int L    = (id >> 4) & 31;
    int rem2 = id >> 9;          // nt*18 + kt
    int kt   = rem2 % KT_ALL;
    int nt   = rem2 / KT_ALL;
    int np   = nt * 16 + (L & 15);
    int g    = (np & 3) * H_DIM + (np >> 2);
    int k    = kt * 32 + (L >> 4) * 16 + j;
    float v = 0.0f;
    if (k < 33)       v = W_ih[g * 33 + k];
    else if (k == 33) v = b_ih[g] + b_hh[g];
    else if (k >= KX) v = W_hh[g * H_DIM + (k - KX)];
    w_pk[id] = (bf16)v;
}

// ---------------------------------------------------------------------------
// Kernel 1: persistent LSTM scan. One workgroup owns 16 batch rows and runs
// all 512 timesteps with the unified A strip [x(64)|h(512)] (bf16, double-
// buffered) and c (f32) in LDS. Wave w computes packed gate columns
// [w*128, w*128+128) via 18x8 WMMAs/step and updates h cols [w*32, w*32+32).
// One __syncthreads per step.
// ---------------------------------------------------------------------------
__global__ __launch_bounds__(512, 1)
void lstm_persist(const float* __restrict__ y,
                  const float* __restrict__ tin,
                  const bf16* __restrict__ w_pk,
                  const float* __restrict__ W_lin,
                  const float* __restrict__ b_lin,
                  float* __restrict__ out)
{
    extern __shared__ char smem[];
    bf16*  s_a   = (bf16*) (smem);                              // [2][BT][KCAT] 36 KB
    float* s_c   = (float*)(smem + 2*BT*KCAT*2);                // [BT][H]       32 KB
    float* s_scr = (float*)(smem + 2*BT*KCAT*2 + BT*H_DIM*4);   // [NWAVE][256]  16 KB

    const int tid  = threadIdx.x;
    const int w    = tid >> 5;       // wave id 0..15
    const int lane = tid & 31;
    const int ra   = lane & 15;      // A-operand row (M)
    const int ks   = lane >> 4;      // K-half select
    const int b0   = blockIdx.x * BT;

    // uniform per-wave weight base + constant per-lane byte offset
    const bf16* pBw     = w_pk + (size_t)w * NT_W * JSTRIDE;
    const int   laneoff = lane * 16;       // elements within a (nt,kt) tile

    // ---- init: strip = 0 (h0 = 0), c0 = 0, x tile for step 0 (t = T-1) ----
    for (int i = tid; i < 2 * BT * KCAT; i += 512) s_a[i] = (bf16)0.0f;
    for (int i = tid; i < BT * H_DIM;  i += 512) s_c[i] = 0.0f;
    __syncthreads();
    {
        const int t_idx = T_LEN - 1;
        for (int e = tid; e < BT * KX; e += 512) {
            int r = e >> 6, col = e & 63;
            float v = 0.0f;
            if (col < 32)       v = y[((size_t)t_idx * B_SZ + b0 + r) * D_IN + col];
            else if (col == 32) v = tin[(size_t)t_idx * B_SZ + b0 + r];
            else if (col == 33) v = 1.0f;
            s_a[r * KCAT + col] = (bf16)v;
        }
    }
    __syncthreads();

    float* scr = s_scr + w * 256;    // wave-private 16x16 f32 scratch
    int cur = 0;

    for (int s = 0; s < T_LEN; ++s) {
        // keep the 512-step loop from hoisting L2-resident weight loads into
        // VGPRs (they must stream; hoisting caused scratch spills)
        asm volatile("" ::: "memory");

        const bf16* acur = s_a + cur * BT * KCAT;
        bf16*       anxt = s_a + (cur ^ 1) * BT * KCAT;

        v8f acc[NT_W] = {};

        // ---- gates = [x|h] @ [W_ihp|W_hh]^T : 18 K-tiles x 8 N-tiles ----
        for (int kt = 0; kt < KT_ALL; ++kt) {
            BFVec a;
            const bf16* ar = acur + ra * KCAT + kt * 32 + ks * 16;
            a.q[0] = *(const uint4*)(ar);
            a.q[1] = *(const uint4*)(ar + 8);

            BFVec b[NT_W];
            const bf16* pk = pBw + (size_t)kt * 512 + laneoff;
            #pragma unroll
            for (int j = 0; j < NT_W; ++j) {
                const uint4* bp = (const uint4*)(pk + j * JSTRIDE);
                b[j].q[0] = bp[0];
                b[j].q[1] = bp[1];
            }
            #pragma unroll
            for (int j = 0; j < NT_W; ++j)
                acc[j] = __builtin_amdgcn_wmma_f32_16x16x32_bf16(
                    false, a.v, false, b[j].v, (short)0, acc[j], false, false);
        }

        // ---- gate nonlinearity + state update (wave-local) ----
        // Acc tile j covers packed cols [w*128 + j*16, +16) = h cols
        // w*32 + j*4 .. +4, gate = packed&3 in order (i,f,g,o).
        #pragma unroll
        for (int j = 0; j < NT_W; ++j) {
            #pragma unroll
            for (int i = 0; i < 8; ++i)
                scr[(i + ks * 8) * 16 + ra] = acc[j][i];   // C/D layout -> scratch
            #pragma unroll
            for (int it = 0; it < 2; ++it) {
                int mm = lane + it * 32;      // 0..63 : 16 rows x 4 h-cols
                int rr = mm & 15;
                int hc = mm >> 4;
                float iv = scr[rr * 16 + hc * 4 + 0];
                float fv = scr[rr * 16 + hc * 4 + 1];
                float gv = scr[rr * 16 + hc * 4 + 2];
                float ov = scr[rr * 16 + hc * 4 + 3];
                int col = w * 32 + j * 4 + hc;
                float c  = s_c[rr * H_DIM + col];
                float cn = sigm_(fv) * c + sigm_(iv) * tanh_(gv);
                s_c[rr * H_DIM + col] = cn;
                anxt[rr * KCAT + KX + col] = (bf16)(sigm_(ov) * tanh_(cn));
            }
        }

        // ---- stage x columns of the next strip ----
        if (s + 1 < T_LEN) {
            const int t_idx = T_LEN - 2 - s;
            for (int e = tid; e < BT * KX; e += 512) {
                int r = e >> 6, col = e & 63;
                float v = 0.0f;
                if (col < 32)       v = y[((size_t)t_idx * B_SZ + b0 + r) * D_IN + col];
                else if (col == 32) v = tin[(size_t)t_idx * B_SZ + b0 + r];
                else if (col == 33) v = 1.0f;
                anxt[r * KCAT + col] = (bf16)v;
            }
        }

        __syncthreads();
        cur ^= 1;
    }

    // ---- final linear: out = h_last @ W_lin^T + b_lin  (16 rows x 32 outs) ----
    const bf16* hfin = s_a + cur * BT * KCAT + KX;
    {
        int r = tid >> 5;    // 0..15
        int o = tid & 31;    // 0..31
        float sum = b_lin[o];
        for (int k = 0; k < H_DIM; ++k)
            sum += (float)hfin[r * KCAT + k] * W_lin[o * H_DIM + k];
        out[(size_t)(b0 + r) * 32 + o] = sum;
    }
}

// ---------------------------------------------------------------------------
extern "C" void kernel_launch(void* const* d_in, const int* in_sizes, int n_in,
                              void* d_out, int out_size, void* d_ws, size_t ws_size,
                              hipStream_t stream)
{
    const float* y     = (const float*)d_in[0];
    const float* tin   = (const float*)d_in[1];
    const float* W_ih  = (const float*)d_in[2];
    const float* W_hh  = (const float*)d_in[3];
    const float* b_ih  = (const float*)d_in[4];
    const float* b_hh  = (const float*)d_in[5];
    const float* W_lin = (const float*)d_in[6];
    const float* b_lin = (const float*)d_in[7];
    float* out = (float*)d_out;

    bf16* w_pk = (bf16*)d_ws;                 // 2048*576*2 = 2.25 MB

    const int totalPack = G_DIM * KCAT;
    pack_weights<<<(totalPack + 255) / 256, 256, 0, stream>>>(
        W_ih, W_hh, b_ih, b_hh, w_pk);

    const size_t lds_bytes = (size_t)2 * BT * KCAT * 2    // [x|h] double-buffer (bf16)
                           + (size_t)BT * H_DIM * 4       // c (f32)
                           + (size_t)NWAVE * 256 * 4;     // per-wave gate scratch
    lstm_persist<<<B_SZ / BT, NWAVE * 32, lds_bytes, stream>>>(
        y, tin, w_pk, W_lin, b_lin, out);
}
